// Attention_70952859729993
// MI455X (gfx1250) — matrix operand order repaired and checked by reading, not compiled, not run on hardware
//
#include <hip/hip_runtime.h>
#include <math.h>

// ---------------------------------------------------------------------------
// Problem constants (from reference)
// ---------------------------------------------------------------------------
#define Bn     4
#define Nn     1024
#define DIMn   1024
#define HEADSn 16
#define DHn    64
#define INNERn 1024
#define TOPKn  256
#define SCALEf 0.125f
#define LNEPSf 1e-5f

typedef __bf16 bf16;
typedef __attribute__((ext_vector_type(16))) __bf16 bf16x16;
typedef __attribute__((ext_vector_type(8)))  float  f32x8;

#define WMMA_BF16(a, b, c) \
  __builtin_amdgcn_wmma_f32_16x16x32_bf16(false, (a), false, (b), (short)0, (c), false, false)

static __device__ __forceinline__ bf16x16 frag2(uint4 lo, uint4 hi) {
  union { uint4 u[2]; bf16x16 v; } x;
  x.u[0] = lo; x.u[1] = hi;
  return x.v;
}
static __device__ __forceinline__ uint4 ldg4(const bf16* p) { return *(const uint4*)p; }

// ---------------------------------------------------------------------------
// fp32 -> bf16 conversion (grid-stride)
// ---------------------------------------------------------------------------
__global__ void cvt_bf16_kernel(const float* __restrict__ src, bf16* __restrict__ dst, int n) {
  int i = blockIdx.x * blockDim.x + threadIdx.x;
  int stride = gridDim.x * blockDim.x;
  for (; i < n; i += stride) dst[i] = (bf16)src[i];
}

__global__ void zero_f32_kernel(float* __restrict__ p, int n) {
  int i = blockIdx.x * blockDim.x + threadIdx.x;
  int stride = gridDim.x * blockDim.x;
  for (; i < n; i += stride) p[i] = 0.0f;
}

// ---------------------------------------------------------------------------
// QKV GEMM: QKV[4096,3072] = Xb[4096,1024] @ Wqkv[3072,1024]^T  (bf16 WMMA)
// Direct load-then-consume form: this is the structure the gfx1250 scheduler
// compiles spill-free (clause'd b128 loads + wmma, compiler 2x unroll).
// Writes Q,K as [B,H,N,DH] bf16, V transposed as [B,H,DH,N] bf16,
// and accumulates fp32 q_mean[4096,64] (mean over heads of Q) via atomics.
// One wave computes a 16x64 tile. 8 waves / block.
// ---------------------------------------------------------------------------
__global__ __launch_bounds__(256)
void qkv_gemm_kernel(const bf16* __restrict__ X,
                     const bf16* __restrict__ W,
                     bf16* __restrict__ Qb,
                     bf16* __restrict__ Kb,
                     bf16* __restrict__ VT,
                     float* __restrict__ qmean) {
  const int lane = threadIdx.x & 31;
  const int wid  = threadIdx.x >> 5;
  const int hl   = lane >> 4;
  const int ln   = lane & 15;
  const int w    = blockIdx.x * 8 + wid;
  const int NT   = (3 * INNERn) / 64;         // 48 n-tiles
  const int mt   = w / NT;
  const int nt   = w % NT;
  const int m0   = mt * 16;
  const int n0   = nt * 64;

  f32x8 acc[4] = {};
  const bf16* arow = X + (size_t)(m0 + ln) * DIMn;

  for (int k0 = 0; k0 < DIMn; k0 += 32) {
    bf16x16 a = frag2(ldg4(arow + k0 + hl * 16), ldg4(arow + k0 + hl * 16 + 8));
#pragma unroll
    for (int c = 0; c < 4; ++c) {
      const bf16* brow = W + (size_t)(n0 + c * 16 + ln) * DIMn + k0 + hl * 16;
      bf16x16 b = frag2(ldg4(brow), ldg4(brow + 8));
      acc[c] = WMMA_BF16(a, b, acc[c]);
    }
  }

  const int region = n0 >> 10;                 // 0=q, 1=k, 2=v
  const int h      = (n0 & 1023) >> 6;         // head index (tile within one head)
#pragma unroll
  for (int c = 0; c < 4; ++c) {
#pragma unroll
    for (int r = 0; r < 8; ++r) {
      float v = acc[c][r];
      int m = m0 + r + 8 * hl;                 // global token row (b*1024 + t)
      int b2 = m >> 10, t = m & 1023;
      int d = c * 16 + ln;
      if (region == 0) {
        Qb[((size_t)(b2 * HEADSn + h) * Nn + t) * DHn + d] = (bf16)v;
        atomicAdd(qmean + (size_t)m * DHn + d, v * 0.0625f);
      } else if (region == 1) {
        Kb[((size_t)(b2 * HEADSn + h) * Nn + t) * DHn + d] = (bf16)v;
      } else {
        VT[((size_t)(b2 * HEADSn + h) * DHn + d) * Nn + t] = (bf16)v;
      }
    }
  }
}

// ---------------------------------------------------------------------------
// Gate scores: LayerNorm(q_mean) -> GELU MLP 64->16->1   (fp32, one thread/token)
// ---------------------------------------------------------------------------
__global__ void gate_score_kernel(const float* __restrict__ qmean,
                                  const float* __restrict__ ln_g, const float* __restrict__ ln_b,
                                  const float* __restrict__ w1,   const float* __restrict__ b1,
                                  const float* __restrict__ w2,   const float* __restrict__ b2,
                                  float* __restrict__ scores) {
  int t = blockIdx.x * blockDim.x + threadIdx.x;
  if (t >= Bn * Nn) return;
  const float* q = qmean + (size_t)t * DHn;
  float mu = 0.f, m2 = 0.f;
  for (int d = 0; d < DHn; ++d) { float v = q[d]; mu += v; m2 += v * v; }
  mu *= (1.0f / DHn);
  float var = m2 * (1.0f / DHn) - mu * mu;
  float inv = rsqrtf(var + LNEPSf);
  float sc = b2[0];
  for (int p = 0; p < 16; ++p) {
    float a = b1[p];
    for (int d = 0; d < DHn; ++d) {
      float H = (q[d] - mu) * inv * ln_g[d] + ln_b[d];
      a += H * w1[p * DHn + d];
    }
    float g = 0.5f * a * (1.0f + erff(a * 0.70710678118654752f));  // exact GELU
    sc += g * w2[p];
  }
  scores[t] = sc / (1.0f + 1e-12f);
}

// ---------------------------------------------------------------------------
// Top-k gate per batch: exact rank with jax.lax.top_k tie-break (lower index wins).
// selected -> sigmoid(score), else sigmoid(-1e9) == 0.
// ---------------------------------------------------------------------------
__global__ void topk_gate_kernel(const float* __restrict__ scores, float* __restrict__ gate) {
  __shared__ float s[Nn];
  int b = blockIdx.x, t = threadIdx.x;
  s[t] = scores[b * Nn + t];
  __syncthreads();
  float st = s[t];
  int cnt = 0;
  for (int j = 0; j < Nn; ++j) {
    float sj = s[j];
    cnt += (sj > st) || (sj == st && j < t);
  }
  float g = 0.0f;
  if (cnt < TOPKn) g = 1.0f / (1.0f + expf(-st));
  gate[b * Nn + t] = g;
}

// ---------------------------------------------------------------------------
// Flash attention per (b,h): S = (Q K^T)*SCALE*g_i*g_j, online softmax, O = P V.
// 8 waves/block, each wave owns 16 query rows; j-tiles of 32 keys.
// P converted C-layout -> A-layout through wave-private LDS.
// ---------------------------------------------------------------------------
__global__ __launch_bounds__(256)
void attention_kernel(const bf16* __restrict__ Qb, const bf16* __restrict__ Kb,
                      const bf16* __restrict__ VT, const float* __restrict__ gate,
                      bf16* __restrict__ Ao) {
  __shared__ __align__(16) bf16 Plds[8 * 16 * 32];   // 8 KB, 1 KB per wave
  const int lane = threadIdx.x & 31;
  const int wid  = threadIdx.x >> 5;
  const int hl   = lane >> 4;
  const int ln   = lane & 15;
  const int idx  = blockIdx.x;
  const int rb   = idx & 7;
  const int h    = (idx >> 3) & 15;
  const int b    = idx >> 7;
  const int m0   = rb * 128 + wid * 16;

  const size_t head = (size_t)(b * HEADSn + h) * Nn * DHn;
  const bf16* Qh  = Qb + head;
  const bf16* Kh  = Kb + head;
  const bf16* Vth = VT + head;                  // [64][1024]
  const float* gb = gate + b * Nn;

  // Q A-fragments (16 rows x 64 dh = two 16x32 fragments)
  const bf16* qrow = Qh + (size_t)(m0 + ln) * DHn;
  bf16x16 qa0 = frag2(ldg4(qrow + hl * 16),      ldg4(qrow + hl * 16 + 8));
  bf16x16 qa1 = frag2(ldg4(qrow + 32 + hl * 16), ldg4(qrow + 40 + hl * 16));

  float grow[8];
#pragma unroll
  for (int r = 0; r < 8; ++r) grow[r] = gb[m0 + r + 8 * hl] * SCALEf;

  float mrun[8], srun[8];
  f32x8 o[4] = {};
#pragma unroll
  for (int r = 0; r < 8; ++r) { mrun[r] = -INFINITY; srun[r] = 0.0f; }

  bf16* pl = Plds + wid * 512;

  for (int j0 = 0; j0 < Nn; j0 += 32) {
    // ---- S tile (16 x 32) = Q @ K^T over dh=64 ----
    f32x8 s0 = {}, s1 = {};
    {
      const bf16* kr = Kh + (size_t)(j0 + ln) * DHn;
      s0 = WMMA_BF16(qa0, frag2(ldg4(kr + hl * 16),      ldg4(kr + hl * 16 + 8)),  s0);
      s0 = WMMA_BF16(qa1, frag2(ldg4(kr + 32 + hl * 16), ldg4(kr + 40 + hl * 16)), s0);
    }
    {
      const bf16* kr = Kh + (size_t)(j0 + 16 + ln) * DHn;
      s1 = WMMA_BF16(qa0, frag2(ldg4(kr + hl * 16),      ldg4(kr + hl * 16 + 8)),  s1);
      s1 = WMMA_BF16(qa1, frag2(ldg4(kr + 32 + hl * 16), ldg4(kr + 40 + hl * 16)), s1);
    }
    // ---- scale + gate (before softmax; gated entries stay 0, not -inf) ----
    float gc0 = gb[j0 + ln], gc1 = gb[j0 + 16 + ln];
    float f[8];
#pragma unroll
    for (int r = 0; r < 8; ++r) {
      s0[r] *= grow[r] * gc0;
      s1[r] *= grow[r] * gc1;
      float tm = fmaxf(s0[r], s1[r]);
#pragma unroll
      for (int msk = 1; msk < 16; msk <<= 1) tm = fmaxf(tm, __shfl_xor(tm, msk, 32));
      float mnew = fmaxf(mrun[r], tm);
      f[r] = __expf(mrun[r] - mnew);
      float p0 = __expf(s0[r] - mnew);
      float p1 = __expf(s1[r] - mnew);
      s0[r] = p0; s1[r] = p1;
      float rs = p0 + p1;
#pragma unroll
      for (int msk = 1; msk < 16; msk <<= 1) rs += __shfl_xor(rs, msk, 32);
      srun[r] = srun[r] * f[r] + rs;
      mrun[r] = mnew;
    }
#pragma unroll
    for (int c = 0; c < 4; ++c)
#pragma unroll
      for (int r = 0; r < 8; ++r) o[c][r] *= f[r];

    // ---- P: C-layout -> LDS (row major 16x32) -> A-layout fragment ----
#pragma unroll
    for (int r = 0; r < 8; ++r) {
      int row = r + 8 * hl;
      pl[row * 32 + ln]      = (bf16)s0[r];
      pl[row * 32 + 16 + ln] = (bf16)s1[r];
    }
    __syncthreads();
    const bf16* pr = pl + (size_t)ln * 32;
    bf16x16 pa = frag2(ldg4(pr + hl * 16), ldg4(pr + hl * 16 + 8));

    // ---- O += P(16x32) @ V_j(32x64), V pre-transposed so B loads are contiguous
#pragma unroll
    for (int c = 0; c < 4; ++c) {
      const bf16* vb = Vth + (size_t)(c * 16 + ln) * Nn + j0 + hl * 16;
      o[c] = WMMA_BF16(pa, frag2(ldg4(vb), ldg4(vb + 8)), o[c]);
    }
    __syncthreads();
  }

  // ---- normalize and write [B,N,H*DH] bf16 ----
#pragma unroll
  for (int r = 0; r < 8; ++r) {
    float inv = 1.0f / srun[r];
    int t = m0 + r + 8 * hl;
    bf16* dst = Ao + ((size_t)(b * Nn + t)) * INNERn + h * DHn + ln;
    dst[0]  = (bf16)(o[0][r] * inv);
    dst[16] = (bf16)(o[1][r] * inv);
    dst[32] = (bf16)(o[2][r] * inv);
    dst[48] = (bf16)(o[3][r] * inv);
  }
}

// ---------------------------------------------------------------------------
// Output GEMM: out[4096,1024] = Ao[4096,1024] @ Wout[1024,1024]^T + b_out (f32 out)
// ---------------------------------------------------------------------------
__global__ __launch_bounds__(256)
void out_gemm_kernel(const bf16* __restrict__ A, const bf16* __restrict__ W,
                     const float* __restrict__ bias, float* __restrict__ out) {
  const int lane = threadIdx.x & 31;
  const int wid  = threadIdx.x >> 5;
  const int hl   = lane >> 4;
  const int ln   = lane & 15;
  const int w    = blockIdx.x * 8 + wid;
  const int NT   = DIMn / 64;                  // 16 n-tiles
  const int mt   = w / NT;
  const int nt   = w % NT;
  const int m0   = mt * 16;
  const int n0   = nt * 64;

  f32x8 acc[4] = {};
  const bf16* arow = A + (size_t)(m0 + ln) * INNERn;

  for (int k0 = 0; k0 < INNERn; k0 += 32) {
    bf16x16 a = frag2(ldg4(arow + k0 + hl * 16), ldg4(arow + k0 + hl * 16 + 8));
#pragma unroll
    for (int c = 0; c < 4; ++c) {
      const bf16* brow = W + (size_t)(n0 + c * 16 + ln) * INNERn + k0 + hl * 16;
      acc[c] = WMMA_BF16(a, frag2(ldg4(brow), ldg4(brow + 8)), acc[c]);
    }
  }

#pragma unroll
  for (int c = 0; c < 4; ++c) {
    int n = n0 + c * 16 + ln;
    float bb = bias[n];
#pragma unroll
    for (int r = 0; r < 8; ++r) {
      int m = m0 + r + 8 * hl;
      out[(size_t)m * DIMn + n] = acc[c][r] + bb;
    }
  }
}

// ---------------------------------------------------------------------------
// Launch
// ---------------------------------------------------------------------------
extern "C" void kernel_launch(void* const* d_in, const int* in_sizes, int n_in,
                              void* d_out, int out_size, void* d_ws, size_t ws_size,
                              hipStream_t stream) {
  (void)in_sizes; (void)n_in; (void)out_size; (void)ws_size;

  const float* x     = (const float*)d_in[0];
  const float* w_qkv = (const float*)d_in[1];
  const float* w_out = (const float*)d_in[2];
  const float* b_out = (const float*)d_in[3];
  const float* ln_g  = (const float*)d_in[4];
  const float* ln_b  = (const float*)d_in[5];
  const float* w1    = (const float*)d_in[6];
  const float* b1    = (const float*)d_in[7];
  const float* w2    = (const float*)d_in[8];
  const float* b2    = (const float*)d_in[9];

  char*  ws  = (char*)d_ws;
  size_t off = 0;
  auto alloc = [&](size_t bytes) -> void* {
    void* p = ws + off;
    off += (bytes + 255) & ~(size_t)255;
    return p;
  };
  bf16*  Xb    = (bf16*)alloc((size_t)Bn * Nn * DIMn * 2);          // 8 MB
  bf16*  Wqkvb = (bf16*)alloc((size_t)3 * INNERn * DIMn * 2);       // 6 MB
  bf16*  Woutb = (bf16*)alloc((size_t)DIMn * INNERn * 2);           // 2 MB
  bf16*  Qb    = (bf16*)alloc((size_t)Bn * HEADSn * Nn * DHn * 2);  // 8 MB
  bf16*  Kb    = (bf16*)alloc((size_t)Bn * HEADSn * Nn * DHn * 2);  // 8 MB
  bf16*  VTb   = (bf16*)alloc((size_t)Bn * HEADSn * DHn * Nn * 2);  // 8 MB
  bf16*  Aob   = (bf16*)alloc((size_t)Bn * Nn * INNERn * 2);        // 8 MB
  float* qmean = (float*)alloc((size_t)Bn * Nn * DHn * 4);          // 1 MB
  float* scoresb = (float*)alloc((size_t)Bn * Nn * 4);
  float* gateb   = (float*)alloc((size_t)Bn * Nn * 4);

  zero_f32_kernel<<<256, 256, 0, stream>>>(qmean, Bn * Nn * DHn);
  cvt_bf16_kernel<<<1024, 256, 0, stream>>>(x,     Xb,    Bn * Nn * DIMn);
  cvt_bf16_kernel<<<768,  256, 0, stream>>>(w_qkv, Wqkvb, 3 * INNERn * DIMn);
  cvt_bf16_kernel<<<256,  256, 0, stream>>>(w_out, Woutb, DIMn * INNERn);

  // (4096/16) * (3072/64) = 12288 wave tiles / 8 waves = 1536 blocks
  qkv_gemm_kernel<<<1536, 256, 0, stream>>>(Xb, Wqkvb, Qb, Kb, VTb, qmean);

  gate_score_kernel<<<(Bn * Nn + 255) / 256, 256, 0, stream>>>(
      qmean, ln_g, ln_b, w1, b1, w2, b2, scoresb);
  topk_gate_kernel<<<Bn, Nn, 0, stream>>>(scoresb, gateb);

  // B*H*(N/128) = 4*16*8 = 512 blocks
  attention_kernel<<<512, 256, 0, stream>>>(Qb, Kb, VTb, gateb, Aob);

  // (4096/16) * (1024/64) = 4096 wave tiles / 8 = 512 blocks
  out_gemm_kernel<<<512, 256, 0, stream>>>(Aob, Woutb, b_out, (float*)d_out);
}
